// LengthRegulator_46016279609817
// MI455X (gfx1250) — matrix operand order repaired and checked by reading, not compiled, not run on hardware
//
#include <hip/hip_runtime.h>
#include <hip/hip_bf16.h>
#include <stdint.h>

// ---- problem sizes (fixed by reference) ----
#define BB 32
#define SS 512
#define DD 256
#define FF 256
#define MM (BB * SS)      // 16384 rows
#define NKSTEP 24         // K = 3*256 = 768 = 24 * 32

// CDNA5 async global->LDS staging (ASYNCcnt path); verified to assemble.
#define USE_ASYNC 1

typedef __attribute__((ext_vector_type(16))) __bf16 v16bf;
typedef __attribute__((ext_vector_type(8)))  float  v8f;

// ---------------------------------------------------------------------------
// fp32 -> bf16 conversion of x (4 elements / thread)
// ---------------------------------------------------------------------------
__global__ __launch_bounds__(256) void k_cvt_x(const float* __restrict__ x,
                                               __bf16* __restrict__ xb) {
  int i = (blockIdx.x * 256 + threadIdx.x) * 4;
  float4 v = *(const float4*)(x + i);
  union { __bf16 h[4]; uint2 u; } p;
  p.h[0] = (__bf16)v.x; p.h[1] = (__bf16)v.y;
  p.h[2] = (__bf16)v.z; p.h[3] = (__bf16)v.w;
  *(uint2*)(xb + i) = p.u;
}

// ---------------------------------------------------------------------------
// Pack conv weights [F, D, K] fp32 into WMMA B-fragment-native bf16 layout:
//   wf[((kstep*16 + ntile)*32 + lane)*16 + j]
// B-frag (32x16): n = lane&15, K = (lane>>4)*16 + j; GEMM K-dim is k-major
// (K_global = tap*256 + c), each kstep covers 32 of it.
// ---------------------------------------------------------------------------
__global__ __launch_bounds__(256) void k_pack_w(const float* __restrict__ w,
                                                __bf16* __restrict__ wf) {
  int e = blockIdx.x * 256 + threadIdx.x;       // < 24*16*32*16 = 196608
  int j     = e & 15;
  int lane  = (e >> 4) & 31;
  int nt    = (e >> 9) & 15;
  int kstep = e >> 13;                          // 0..23
  int f  = nt * 16 + (lane & 15);
  int c  = ((kstep & 7) << 5) + ((lane >> 4) << 4) + j;   // channel 0..255
  int ks = kstep >> 3;                                     // tap 0..2
  wf[e] = (__bf16)w[f * (DD * 3) + c * 3 + ks];
}

// ---------------------------------------------------------------------------
// Conv-as-GEMM, WMMA bf16: H[m,f] = sum_{k,c} A[b,s+k-1,c] * W[f,c,k] + bias[f]
// Block: 256 thr (8 waves). Block tile M=64, N=256. Wave: 16 rows x 128 cols
// (8 accumulators). The conv taps overlap, so the WHOLE halo'd A slab
// (66 rows x 256 ch bf16 = 33 KB) is staged into LDS once (async), then the
// K loop runs barrier-free with an explicit 2-stage register pipeline on the
// B fragments so loads overlap the WMMA stream.
// ---------------------------------------------------------------------------
__global__ __launch_bounds__(256) void k_conv_gemm(const __bf16* __restrict__ A,
                                                   const uint4* __restrict__ wf,
                                                   const float* __restrict__ bias,
                                                   float* __restrict__ H) {
  __shared__ unsigned Alds[66 * 128];   // 66 rows x 256 bf16 = 33,792 B

  const int tid   = threadIdx.x;
  const int lane  = tid & 31;
  const int w     = tid >> 5;           // wave 0..7
  const int hi    = lane >> 4;          // lane half
  const int mSub  = w & 3;              // M sub-tile (16 rows each)
  const int nHalf = w >> 2;             // N half (128 cols each)
  const int mBase = blockIdx.x * 64;    // block fully inside one batch (64|512)
  const int bS    = mBase >> 9;
  const int s0    = mBase & 511;

  // ---- stage full halo'd A slab: rows s0-1 .. s0+64, zeros outside batch ----
  for (int idx = tid; idx < 66 * 32; idx += 256) {   // uint4 granules
    const int row = idx >> 5;
    const int q   = idx & 31;
    unsigned* dstu = &Alds[row * 128 + q * 4];
    const int s = s0 - 1 + row;
    if (s >= 0 && s < SS) {
      const __bf16* gsrc = A + ((size_t)(bS << 9) + s) * DD + q * 8;
#if USE_ASYNC
      unsigned laddr = (unsigned)(size_t)dstu;   // low 32 bits of LDS aperture
      asm volatile("global_load_async_to_lds_b128 %0, %1, off"
                   :: "v"(laddr), "v"(gsrc) : "memory");
#else
      *(uint4*)dstu = *(const uint4*)gsrc;
#endif
    } else {
      *(uint4*)dstu = make_uint4(0, 0, 0, 0);
    }
  }
#if USE_ASYNC
  asm volatile("s_wait_asynccnt 0x0" ::: "memory");
#endif
  __syncthreads();

  // ---- fragment helpers ----
  struct AFrag { union { v16bf v; uint4 q[2]; }; };
  struct BFrag { union { v16bf v; uint4 q[2]; }; };

  const int rA = mSub * 16 + (lane & 15);      // LDS row (before +tap)

  auto loadA = [&](int kstep) -> AFrag {
    const int ks = kstep >> 3;                 // tap 0..2 -> row shift
    const int c0 = (kstep & 7) << 5;           // channel window base
    const uint4* arow = (const uint4*)&Alds[(rA + ks) * 128 + (c0 >> 1)];
    AFrag af;
    af.q[0] = arow[2 * hi];        // K {hi*16 .. hi*16+7}
    af.q[1] = arow[2 * hi + 1];    // K {hi*16+8 .. hi*16+15}
    return af;
  };

  auto loadB = [&](BFrag* bf, int kstep) {
    const uint4* bbase =
        wf + ((size_t)(kstep * 16 + nHalf * 8) * 32 + lane) * 2;
#pragma unroll
    for (int nt = 0; nt < 8; ++nt) {
      bf[nt].q[0] = bbase[nt * 64];
      bf[nt].q[1] = bbase[nt * 64 + 1];
    }
  };

  v8f acc[8] = {};
  auto mm = [&](const AFrag& a, BFrag* bf) {
#pragma unroll
    for (int nt = 0; nt < 8; ++nt) {
      acc[nt] = __builtin_amdgcn_wmma_f32_16x16x32_bf16(
          false, a.v, false, bf[nt].v, (short)0, acc[nt], false, false);
    }
  };

  // ---- barrier-free K loop, 2-stage B pipeline ----
  BFrag b0[8], b1[8];
  loadB(b0, 0);
  for (int kk = 0; kk < NKSTEP; kk += 2) {
    AFrag a0 = loadA(kk);
    loadB(b1, kk + 1);                 // consumers a phase away -> stays hoisted
    mm(a0, b0);
    AFrag a1 = loadA(kk + 1);
    if (kk + 2 < NKSTEP) loadB(b0, kk + 2);
    mm(a1, b1);
  }

  // ---- epilogue: C/D layout — VGPR e holds row (e + hi*8), col = lane&15 ----
  const int row0 = mBase + mSub * 16 + hi * 8;
#pragma unroll
  for (int nt = 0; nt < 8; ++nt) {
    const int col = nHalf * 128 + nt * 16 + (lane & 15);
    const float bc = bias[col];
    float* hp = H + (size_t)row0 * FF + col;
#pragma unroll
    for (int e = 0; e < 8; ++e) hp[(size_t)e * FF] = acc[nt][e] + bc;
  }
}

// ---------------------------------------------------------------------------
// LayerNorm(256) + ReLU -> bf16 (feeds second conv GEMM)
// ---------------------------------------------------------------------------
__global__ __launch_bounds__(256) void k_ln_relu(const float* __restrict__ H,
                                                 const float* __restrict__ g,
                                                 const float* __restrict__ bta,
                                                 __bf16* __restrict__ out) {
  __shared__ float red[256];
  const int row = blockIdx.x;
  const int f = threadIdx.x;
  const float v = H[(size_t)row * FF + f];
  red[f] = v; __syncthreads();
  for (int off = 128; off > 0; off >>= 1) { if (f < off) red[f] += red[f + off]; __syncthreads(); }
  const float mu = red[0] * (1.0f / FF);
  __syncthreads();
  const float d = v - mu;
  red[f] = d * d; __syncthreads();
  for (int off = 128; off > 0; off >>= 1) { if (f < off) red[f] += red[f + off]; __syncthreads(); }
  const float var = red[0] * (1.0f / FF);
  float y = d * rsqrtf(var + 1e-5f) * g[f] + bta[f];
  y = fmaxf(y, 0.0f);
  out[(size_t)row * FF + f] = (__bf16)y;
}

// ---------------------------------------------------------------------------
// LayerNorm(256) + ReLU + dot(lin_w) + lin_b + ReLU -> predicted_len[row]
// ---------------------------------------------------------------------------
__global__ __launch_bounds__(256) void k_ln_relu_dot(const float* __restrict__ H,
                                                     const float* __restrict__ g,
                                                     const float* __restrict__ bta,
                                                     const float* __restrict__ lw,
                                                     const float* __restrict__ lb,
                                                     float* __restrict__ pred) {
  __shared__ float red[256];
  const int row = blockIdx.x;
  const int f = threadIdx.x;
  const float v = H[(size_t)row * FF + f];
  red[f] = v; __syncthreads();
  for (int off = 128; off > 0; off >>= 1) { if (f < off) red[f] += red[f + off]; __syncthreads(); }
  const float mu = red[0] * (1.0f / FF);
  __syncthreads();
  const float d = v - mu;
  red[f] = d * d; __syncthreads();
  for (int off = 128; off > 0; off >>= 1) { if (f < off) red[f] += red[f + off]; __syncthreads(); }
  const float var = red[0] * (1.0f / FF);
  __syncthreads();
  float y = fmaxf(d * rsqrtf(var + 1e-5f) * g[f] + bta[f], 0.0f);
  red[f] = y * lw[f]; __syncthreads();
  for (int off = 128; off > 0; off >>= 1) { if (f < off) red[f] += red[f + off]; __syncthreads(); }
  if (f == 0) pred[row] = fmaxf(red[0] + lb[0], 0.0f);
}

// ---------------------------------------------------------------------------
// Per-batch inclusive cumsum of round(target) -> ends[b][s]
// ---------------------------------------------------------------------------
__global__ __launch_bounds__(512) void k_cumsum(const float* __restrict__ target,
                                                float* __restrict__ ends) {
  __shared__ float s[512];
  const int b = blockIdx.x, t = threadIdx.x;
  s[t] = roundf(target[b * SS + t]);
  __syncthreads();
  for (int off = 1; off < 512; off <<= 1) {
    float v = (t >= off) ? s[t - off] : 0.0f;
    __syncthreads();
    s[t] += v;
    __syncthreads();
  }
  ends[b * SS + t] = s[t];
}

// ---------------------------------------------------------------------------
// Length regulate: out[b, t, :] = x[b, s(t), :]  (zeros past batch total)
// s(t) = first index with ends[s] > t, branchless binary search.
// ---------------------------------------------------------------------------
__global__ __launch_bounds__(64) void k_regulate(const float* __restrict__ x,
                                                 const float* __restrict__ ends,
                                                 float* __restrict__ out, int L) {
  const int t = blockIdx.x, b = blockIdx.y;
  const float* eb = ends + b * SS;
  const float tf = (float)t;
  float4 v = make_float4(0.0f, 0.0f, 0.0f, 0.0f);
  if (tf < eb[SS - 1]) {
    int s = 0;
#pragma unroll
    for (int step = 256; step; step >>= 1) {
      int c = s + step;
      if (c <= SS && eb[c - 1] <= tf) s = c;
    }
    v = ((const float4*)(x + ((size_t)b * SS + s) * DD))[threadIdx.x];
  }
  ((float4*)(out + ((size_t)b * L + t) * DD))[threadIdx.x] = v;
}

// ---------------------------------------------------------------------------
extern "C" void kernel_launch(void* const* d_in, const int* in_sizes, int n_in,
                              void* d_out, int out_size, void* d_ws, size_t ws_size,
                              hipStream_t stream) {
  const float* x      = (const float*)d_in[0];
  const float* target = (const float*)d_in[1];
  const float* w1     = (const float*)d_in[2];
  const float* b1     = (const float*)d_in[3];
  const float* w2     = (const float*)d_in[4];
  const float* b2     = (const float*)d_in[5];
  const float* g1     = (const float*)d_in[6];
  const float* be1    = (const float*)d_in[7];
  const float* g2     = (const float*)d_in[8];
  const float* be2    = (const float*)d_in[9];
  const float* lw     = (const float*)d_in[10];
  const float* lb     = (const float*)d_in[11];
  (void)in_sizes; (void)n_in; (void)ws_size;

  // L recovered from out_size (expand_max_len lives on device; can't read it
  // host-side under graph capture): out = B*L*D floats + B*S floats.
  const int L = (out_size - BB * SS) / (BB * DD);

  // workspace carve-up (~34.4 MB total)
  char* ws = (char*)d_ws;
  __bf16* xbf  = (__bf16*)(ws + 0);           //  8,388,608 B
  __bf16* hbf  = (__bf16*)(ws + 8388608);     //  8,388,608 B
  __bf16* wf1  = (__bf16*)(ws + 16777216);    //    393,216 B
  __bf16* wf2  = (__bf16*)(ws + 17170432);    //    393,216 B
  float*  ends = (float*) (ws + 17563648);    //     65,536 B
  float*  Hf   = (float*) (ws + 17629184);    // 16,777,216 B

  float* out_main = (float*)d_out;
  float* pred     = out_main + (size_t)BB * L * DD;

  k_cvt_x<<<4096, 256, 0, stream>>>(x, xbf);
  k_pack_w<<<768, 256, 0, stream>>>(w1, wf1);
  k_pack_w<<<768, 256, 0, stream>>>(w2, wf2);

  k_conv_gemm<<<MM / 64, 256, 0, stream>>>(xbf, (const uint4*)wf1, b1, Hf);
  k_ln_relu<<<MM, 256, 0, stream>>>(Hf, g1, be1, hbf);
  k_conv_gemm<<<MM / 64, 256, 0, stream>>>(hbf, (const uint4*)wf2, b2, Hf);
  k_ln_relu_dot<<<MM, 256, 0, stream>>>(Hf, g2, be2, lw, lb, pred);

  k_cumsum<<<BB, 512, 0, stream>>>(target, ends);
  k_regulate<<<dim3((unsigned)L, BB), 64, 0, stream>>>(x, ends, out_main, L);
}
